// HyperbolicMSCNN_2989297238178
// MI455X (gfx1250) — matrix operand order repaired and checked by reading, not compiled, not run on hardware
//
#include <hip/hip_runtime.h>
#include <cmath>
#include <cstdint>

typedef __attribute__((ext_vector_type(16))) _Float16 v16h;
typedef __attribute__((ext_vector_type(8)))  float    v8f;

#define EPSF  1e-15f
#define ONE_M 0.9999999f

// ---------------------------------------------------------------- reductions
__device__ inline float wave_sum(float v) {
#pragma unroll
  for (int o = 16; o > 0; o >>= 1) v += __shfl_xor(v, o, 32);
  return v;
}

__device__ inline float block_sum(float v, float* red) {
  const int t = threadIdx.x;
  red[t] = v;
  __syncthreads();
  for (int s = 128; s > 0; s >>= 1) {
    if (t < s) red[t] += red[t + s];
    __syncthreads();
  }
  float rv = red[0];
  __syncthreads();
  return rv;
}

// --------------------------------------------------------------- zero fill
__global__ void k_fill(float* __restrict__ p, int n) {
  int i = blockIdx.x * blockDim.x + threadIdx.x;
  if (i < n) p[i] = 0.f;
}

// ----------------------- logmap0 per pixel, written into halo-padded NHWC
__global__ void k_logmap(const float* __restrict__ x, float* __restrict__ v,
                         int nPix, int C, int H, int W, int pad, int nchw) {
  int p = blockIdx.x * blockDim.x + threadIdx.x;
  if (p >= nPix) return;
  const int Hp = H + 2 * pad, Wp = W + 2 * pad;
  int b = p / (H * W);
  int yx = p - b * H * W;
  int y = yx / W, xq = yx - (yx / W) * W;
  float* vb = v + (((size_t)b * Hp + y + pad) * Wp + (xq + pad)) * C;
  float ssq = 0.f;
  if (nchw) {
    const float* base = x + (size_t)b * C * H * W + yx;
    for (int c = 0; c < C; ++c) { float t = base[(size_t)c * H * W]; ssq += t * t; }
    float n = sqrtf(fmaxf(ssq, EPSF));
    float f = atanhf(fminf(n, ONE_M)) / n;
    for (int c = 0; c < C; ++c) vb[c] = f * base[(size_t)c * H * W];
  } else {
    const float* base = x + (size_t)p * C;
    for (int c = 0; c < C; ++c) { float t = base[c]; ssq += t * t; }
    float n = sqrtf(fmaxf(ssq, EPSF));
    float f = atanhf(fminf(n, ONE_M)) / n;
    for (int c = 0; c < C; ++c) vb[c] = f * base[c];
  }
}

// ------------- per-output-row expmap scale + cx2 (halo makes it branchless)
__global__ void k_rowstats(const float* __restrict__ v, float* __restrict__ rowscale,
                           float* __restrict__ cx2, int B, int H, int W, int Cin,
                           int KH, int pad, float scale) {
  int m = blockIdx.x * blockDim.x + threadIdx.x;
  int M = B * H * W;
  if (m >= M) return;
  const int Hp = H + 2 * pad, Wp = W + 2 * pad;
  int b = m / (H * W);
  int rr = m - b * H * W;
  int y = rr / W, xx = rr - (rr / W) * W;
  const float* base = v + (((size_t)b * Hp + y) * Wp + xx) * Cin;
  float ssq = 0.f;
  for (int ky = 0; ky < KH; ++ky)
    for (int kx = 0; kx < KH; ++kx) {
      const float* t0 = base + ((size_t)ky * Wp + kx) * Cin;
      for (int c = 0; c < Cin; ++c) { float t = t0[c]; ssq += t * t; }
    }
  float nw = sqrtf(fmaxf(scale * scale * ssq, EPSF));
  float th = tanhf(nw);
  rowscale[m] = scale * th / nw;
  cx2[m] = th * th;
}

// ------------------------------------------- z -> unit columns (f16, padded)
__global__ void k_zunit(const float* __restrict__ z, _Float16* __restrict__ zu,
                        float* __restrict__ znorm, int K, int Kp, int N) {
  int n = blockIdx.x * blockDim.x + threadIdx.x;
  if (n >= N) return;
  float ssq = 0.f;
  for (int k = 0; k < K; ++k) { float t = z[(size_t)k * N + n]; ssq += t * t; }
  float nrm = sqrtf(fmaxf(ssq, EPSF));
  znorm[n] = nrm;
  float inv = 1.f / nrm;
  for (int k = 0; k < Kp; ++k)
    zu[(size_t)k * N + n] = (k < K) ? (_Float16)(z[(size_t)k * N + n] * inv)
                                    : (_Float16)0.f;
}

// ------------- im2col delta table (padded coords): k -> (ky*Wp+kx)*Cin + c
// entries k in [K, Kp) get delta 0: their zu rows are zero, so A is don't-care.
__global__ void k_mktab(int* __restrict__ tab, int K, int Kp, int KH, int Wp,
                        int Cin) {
  int k = blockIdx.x * blockDim.x + threadIdx.x;
  if (k >= Kp) return;
  int d = 0;
  if (k < K) {
    int KK = KH * KH;
    int c = k / KK, rr = k - c * KK;
    int ky = rr / KH, kx = rr - (rr / KH) * KH;
    d = (ky * Wp + kx) * Cin + c;
  }
  tab[k] = d;
}

// ---------------------------------------------- fused im2col + WMMA GEMM
// BM=128 BN=32 BK=32, 8 waves in 4x2 grid; each wave owns a 32x16 C tile
// (2 accumulators sharing one B fragment -> 2 v_wmma per K step).
// B tile: global->LDS via global_load_async_to_lds_b128 (ASYNCcnt).
// A gather: halo-padded source + LDS delta table -> zero branches, zero masks.
#define MAXK 3136
__global__ __launch_bounds__(256)
void k_conv_gemm(const float* __restrict__ v, const float* __restrict__ rowscale,
                 const _Float16* __restrict__ zu, const int* __restrict__ tab,
                 float* __restrict__ P,
                 int H, int W, int Hp, int Wp, int Cin,
                 int Kp, int N) {
  __shared__ __align__(16) _Float16 ldsA[128 * 32];   // [row][k]
  __shared__ __align__(16) _Float16 ldsB[32 * 32];    // [k][n]
  __shared__ __align__(16) int      ldsTab[MAXK];
  const int tid  = threadIdx.x;
  const int lane = tid & 31;
  const int wave = tid >> 5;
  const int wm = wave >> 1;            // 0..3 -> 32-row slice
  const int wn = wave & 1;             // 0..1 -> 16-col slice
  const int bm = blockIdx.x * 128;
  const int bn = blockIdx.y * 32;

  // stage im2col delta table into LDS (one time)
  for (int i = tid; i < Kp; i += 256) ldsTab[i] = tab[i];

  // A loader: 2 threads per row, 16 halves each
  const int lrow = tid >> 1;           // 0..127
  const int lkg  = (tid & 1) * 16;     // 0 or 16
  const int m    = bm + lrow;
  const int b    = m / (H * W);
  const int ryx  = m - b * H * W;
  const int py   = ryx / W;
  const int px   = ryx - py * W;
  const float rs = rowscale[m];
  const int ibase = ((b * Hp + py) * Wp + px) * Cin;   // padded-coord base

  // B loader (async): threads 0..127, one b128 (8 halves) each; hoisted ptr
  const int bk = tid >> 2;             // 0..31 (k row)
  const int bc = (tid & 3) * 8;        // chunk of 8 halves
  const _Float16* gB = zu + (size_t)bk * N + bn + bc;

  __syncthreads();                     // table visible to all waves

  v8f acc0 = {}, acc1 = {};
  for (int k0 = 0; k0 < Kp; k0 += 32) {
    // ---- B tile: global -> LDS, asynchronous (no VGPR round trip)
    if (tid < 128) {
      unsigned ldsOff = (unsigned)(uintptr_t)&ldsB[bk * 32 + bc];
      unsigned long long ga = (unsigned long long)(uintptr_t)gB;
      asm volatile("global_load_async_to_lds_b128 %0, %1, off"
                   :: "v"(ldsOff), "v"(ga) : "memory");
    }
    gB += (size_t)32 * N;

    // ---- A tile: batched table reads, batched gathers, packed f16 stores
    int d[16];
#pragma unroll
    for (int j = 0; j < 16; ++j) d[j] = ldsTab[k0 + lkg + j];
    float t[16];
#pragma unroll
    for (int j = 0; j < 16; ++j) t[j] = v[ibase + d[j]];
#pragma unroll
    for (int j = 0; j < 16; j += 2) {
      _Float16 h0 = (_Float16)(rs * t[j]);
      _Float16 h1 = (_Float16)(rs * t[j + 1]);
      unsigned pk = (unsigned)__builtin_bit_cast(unsigned short, h0) |
                    ((unsigned)__builtin_bit_cast(unsigned short, h1) << 16);
      *(unsigned*)&ldsA[lrow * 32 + lkg + j] = pk;
    }
    if (tid == 0 && k0 + 64 <= Kp)
      __builtin_prefetch(&zu[(size_t)(k0 + 32) * N + bn], 0, 1);
    asm volatile("s_wait_asynccnt 0x0" ::: "memory");
    __syncthreads();

    // ---- fragments per ISA 7.12.2 (wave32)
    union { v16h h; unsigned u[8]; } fa0, fa1, fb;
    const int hi = lane >> 4;
    const unsigned* pa0 = (const unsigned*)(ldsA + (wm * 32 + (lane & 15)) * 32);
    const unsigned* pa1 = (const unsigned*)(ldsA + (wm * 32 + 16 + (lane & 15)) * 32);
#pragma unroll
    for (int q = 0; q < 4; ++q) {
      fa0.u[q]     = pa0[hi * 4 + q];          // K = hi*8 + 2q
      fa0.u[4 + q] = pa0[8 + hi * 4 + q];      // K = 16 + hi*8 + 2q
      fa1.u[q]     = pa1[hi * 4 + q];
      fa1.u[4 + q] = pa1[8 + hi * 4 + q];
    }
    const unsigned short* bs = (const unsigned short*)ldsB;
    const int nn = wn * 16 + (lane & 15);
    const int kb = hi * 8;
#pragma unroll
    for (int q = 0; q < 8; ++q)
      fb.u[q] = (unsigned)bs[(kb + 2 * q + ((q & 4) ? 8 : 0)) * 32 + nn] |
                ((unsigned)bs[(kb + 2 * q + ((q & 4) ? 8 : 0) + 1) * 32 + nn] << 16);

    acc0 = __builtin_amdgcn_wmma_f32_16x16x32_f16(false, fa0.h, false, fb.h,
                                                  (short)0, acc0, false, false);
    acc1 = __builtin_amdgcn_wmma_f32_16x16x32_f16(false, fa1.h, false, fb.h,
                                                  (short)0, acc1, false, false);
    __syncthreads();
  }
  const int n_out = bn + wn * 16 + (lane & 15);
  const int mb    = bm + wm * 32 + (lane >> 4) * 8;
#pragma unroll
  for (int q = 0; q < 8; ++q) {
    P[(size_t)(mb + q) * N + n_out]      = acc0[q];
    P[(size_t)(mb + 16 + q) * N + n_out] = acc1[q];
  }
}

// -------------- MLR -> sinh -> ball projection -> hrelu, in place, wave/row
__global__ __launch_bounds__(256)
void k_epilogue(float* __restrict__ P, const float* __restrict__ cx2,
                const float* __restrict__ znorm, const float* __restrict__ r,
                int M, int N) {
  const int lane = threadIdx.x & 31;
  const int wave = threadIdx.x >> 5;
  const int m = blockIdx.x * 8 + wave;
  if (m >= M) return;
  const int npl = N >> 5;
  const float c2 = cx2[m];
  const float inv_den = 1.f / fmaxf(1.f - c2, EPSF);
  float yv[4];
  float s2 = 0.f;
  for (int i = 0; i < npl; ++i) {
    int n = lane + i * 32;
    float drcr = 2.f * r[n];
    float pn = P[(size_t)m * N + n];
    float num = 2.f * pn * coshf(drcr) - (1.f + c2) * sinhf(drcr);
    float mlr = 2.f * znorm[n] * asinhf(num * inv_den);
    float y = sinhf(mlr);
    yv[i] = y;
    s2 += y * y;
  }
  s2 = wave_sum(s2);
  float proj = 1.f / (1.f + sqrtf(1.f + s2));
  float s2p = 0.f;
  for (int i = 0; i < npl; ++i) { yv[i] *= proj; s2p += yv[i] * yv[i]; }
  s2p = wave_sum(s2p);
  float nrm = sqrtf(fmaxf(s2p, EPSF));
  float f = atanhf(fminf(nrm, ONE_M)) / nrm;
  float p2 = 0.f;
  for (int i = 0; i < npl; ++i) { yv[i] = f * fmaxf(yv[i], 0.f); p2 += yv[i] * yv[i]; }
  p2 = wave_sum(p2);
  float np = sqrtf(fmaxf(p2, EPSF));
  float g = tanhf(np) / np;
  for (int i = 0; i < npl; ++i)
    P[(size_t)m * N + lane + i * 32] = g * yv[i];
}

// ---------------------- tangent-space 2x2 max pool (NHWC), wave/output-pixel
__global__ __launch_bounds__(256)
void k_maxpool(const float* __restrict__ a, float* __restrict__ o,
               int B, int H, int W, int C) {
  const int lane = threadIdx.x & 31;
  const int wave = threadIdx.x >> 5;
  const int Ho = H >> 1, Wo = W >> 1;
  const int q = blockIdx.x * 8 + wave;
  if (q >= B * Ho * Wo) return;
  int b = q / (Ho * Wo);
  int rr = q - b * Ho * Wo;
  int yo = rr / Wo, xo = rr - (rr / Wo) * Wo;
  const int cpl = C >> 5;
  float tv[4];
  for (int i = 0; i < cpl; ++i) tv[i] = -1e30f;
  for (int d = 0; d < 4; ++d) {
    int iy = yo * 2 + (d >> 1);
    int ix = xo * 2 + (d & 1);
    const float* base = a + (((size_t)b * H + iy) * W + ix) * C;
    float av[4];
    float ssq = 0.f;
    for (int i = 0; i < cpl; ++i) { av[i] = base[lane + i * 32]; ssq += av[i] * av[i]; }
    ssq = wave_sum(ssq);
    float n = sqrtf(fmaxf(ssq, EPSF));
    float f = atanhf(fminf(n, ONE_M)) / n;
    for (int i = 0; i < cpl; ++i) tv[i] = fmaxf(tv[i], f * av[i]);
  }
  float s = 0.f;
  for (int i = 0; i < cpl; ++i) s += tv[i] * tv[i];
  s = wave_sum(s);
  float nn = sqrtf(fmaxf(s, EPSF));
  float g = tanhf(nn) / nn;
  float* ob = o + (((size_t)b * Ho + yo) * Wo + xo) * C;
  for (int i = 0; i < cpl; ++i) ob[lane + i * 32] = g * tv[i];
}

// --- adaptive_avg(.,8).mean(2,3) == global spatial mean -> flat (16,352)
__global__ void k_flat(const float* __restrict__ x1, const float* __restrict__ x2,
                       const float* __restrict__ x3, const float* __restrict__ x4,
                       float* __restrict__ flat) {
  int idx = blockIdx.x * blockDim.x + threadIdx.x;
  if (idx >= 16 * 352) return;
  int bq = idx / 352, cg = idx - bq * 352;
  const float* src; int C, HW, c;
  if (cg < 32)       { src = x1; C = 32;  HW = 64 * 64; c = cg; }
  else if (cg < 96)  { src = x2; C = 64;  HW = 32 * 32; c = cg - 32; }
  else if (cg < 224) { src = x3; C = 128; HW = 16 * 16; c = cg - 96; }
  else               { src = x4; C = 128; HW = 8 * 8;   c = cg - 224; }
  const float* base = src + (size_t)bq * HW * C + c;
  float s = 0.f;
  for (int p = 0; p < HW; ++p) s += base[(size_t)p * C];
  flat[idx] = s / (float)HW;
}

// ------------------------- poincare_fc (+ optional hrelu), 1 block per batch
__global__ __launch_bounds__(256)
void k_fc(const float* __restrict__ X, const _Float16* __restrict__ zu,
          const float* __restrict__ znorm, const float* __restrict__ r,
          float* __restrict__ O, int Nin, int Nout, int relu) {
  __shared__ float xs[512];
  __shared__ float red[256];
  const int t = threadIdx.x;
  const int bq = blockIdx.x;
  float ssq = 0.f;
  for (int k = t; k < Nin; k += 256) {
    float xv = X[(size_t)bq * Nin + k];
    xs[k] = xv;
    ssq += xv * xv;
  }
  float c2 = block_sum(ssq, red);
  float inv_den = 1.f / fmaxf(1.f - c2, EPSF);
  float yl[4];
  float y2 = 0.f;
  int idx = 0;
  for (int n = t; n < Nout; n += 256, ++idx) {
    float dot = 0.f;
    for (int k = 0; k < Nin; ++k) dot += xs[k] * (float)zu[(size_t)k * Nout + n];
    float drcr = 2.f * r[n];
    float num = 2.f * dot * coshf(drcr) - (1.f + c2) * sinhf(drcr);
    float mlr = 2.f * znorm[n] * asinhf(num * inv_den);
    float y = sinhf(mlr);
    yl[idx] = y;
    y2 += y * y;
  }
  float s2 = block_sum(y2, red);
  float proj = 1.f / (1.f + sqrtf(1.f + s2));
  if (!relu) {
    idx = 0;
    for (int n = t; n < Nout; n += 256, ++idx) O[(size_t)bq * Nout + n] = yl[idx] * proj;
    return;
  }
  float s2p = 0.f;
  idx = 0;
  for (int n = t; n < Nout; n += 256, ++idx) { yl[idx] *= proj; s2p += yl[idx] * yl[idx]; }
  s2p = block_sum(s2p, red);
  float nrm = sqrtf(fmaxf(s2p, EPSF));
  float f = atanhf(fminf(nrm, ONE_M)) / nrm;
  float p2 = 0.f;
  idx = 0;
  for (int n = t; n < Nout; n += 256, ++idx) {
    yl[idx] = f * fmaxf(yl[idx], 0.f);
    p2 += yl[idx] * yl[idx];
  }
  p2 = block_sum(p2, red);
  float np = sqrtf(fmaxf(p2, EPSF));
  float g = tanhf(np) / np;
  idx = 0;
  for (int n = t; n < Nout; n += 256, ++idx) O[(size_t)bq * Nout + n] = g * yl[idx];
}

// ==========================================================================
extern "C" void kernel_launch(void* const* d_in, const int* in_sizes, int n_in,
                              void* d_out, int out_size, void* d_ws, size_t ws_size,
                              hipStream_t stream) {
  const float* x   = (const float*)d_in[0];
  const float* z1  = (const float*)d_in[1];
  const float* b1  = (const float*)d_in[2];
  const float* z2  = (const float*)d_in[3];
  const float* b2  = (const float*)d_in[4];
  const float* z3  = (const float*)d_in[5];
  const float* b3  = (const float*)d_in[6];
  const float* z4  = (const float*)d_in[7];
  const float* b4  = (const float*)d_in[8];
  const float* zf1 = (const float*)d_in[9];
  const float* bf1 = (const float*)d_in[10];
  const float* zf2 = (const float*)d_in[11];
  const float* bf2 = (const float*)d_in[12];
  (void)in_sizes; (void)n_in; (void)out_size; (void)ws_size;

  char* base = (char*)d_ws;
  size_t off = 0;
  auto alloc = [&](size_t bytes) -> void* {
    off = (off + 255) & ~(size_t)255;
    void* p = base + off;
    off += bytes;
    return p;
  };
  float* v        = (float*)alloc((size_t)2367488 * 4);  // max padded tangent (L2: 16*68*68*32)
  float* P        = (float*)alloc((size_t)8388608 * 4);
  float* rowscale = (float*)alloc(262144 * 4);
  float* cx2      = (float*)alloc(262144 * 4);
  float* x1p      = (float*)alloc(2097152 * 4);   // (16,64,64,32)  NHWC
  float* x2p      = (float*)alloc(1048576 * 4);   // (16,32,32,64)
  float* x3p      = (float*)alloc(524288 * 4);    // (16,16,16,128)
  float* x4p      = (float*)alloc(131072 * 4);    // (16,8,8,128)
  _Float16* zu1  = (_Float16*)alloc(32 * 32 * 2);
  _Float16* zu2  = (_Float16*)alloc(800 * 64 * 2);
  _Float16* zu3  = (_Float16*)alloc((size_t)3136 * 128 * 2);
  _Float16* zu4  = (_Float16*)alloc((size_t)1152 * 128 * 2);
  _Float16* zuf1 = (_Float16*)alloc((size_t)352 * 512 * 2);
  _Float16* zuf2 = (_Float16*)alloc((size_t)512 * 1000 * 2);
  float* zn1  = (float*)alloc(32 * 4);
  float* zn2  = (float*)alloc(64 * 4);
  float* zn3  = (float*)alloc(128 * 4);
  float* zn4  = (float*)alloc(128 * 4);
  float* znf1 = (float*)alloc(512 * 4);
  float* znf2 = (float*)alloc(1000 * 4);
  int* t1 = (int*)alloc(32 * 4);
  int* t2 = (int*)alloc(800 * 4);
  int* t3 = (int*)alloc(3136 * 4);
  int* t4 = (int*)alloc(1152 * 4);
  float* flat = (float*)alloc(16 * 352 * 4);
  float* hbuf = (float*)alloc(16 * 512 * 4);

  auto betaf = [](double a, double b) {
    return std::exp(std::lgamma(a) + std::lgamma(b) - std::lgamma(a + b));
  };
  float sc1 = (float)(betaf(27.0 / 2.0, 0.5)   / betaf(3.0 / 2.0, 0.5));
  float sc2 = (float)(betaf(800.0 / 2.0, 0.5)  / betaf(32.0 / 2.0, 0.5));
  float sc3 = (float)(betaf(3136.0 / 2.0, 0.5) / betaf(64.0 / 2.0, 0.5));
  float sc4 = (float)(betaf(1152.0 / 2.0, 0.5) / betaf(128.0 / 2.0, 0.5));

  k_zunit<<<1, 256, 0, stream>>>(z1, zu1, zn1, 27, 32, 32);
  k_zunit<<<1, 256, 0, stream>>>(z2, zu2, zn2, 800, 800, 64);
  k_zunit<<<1, 256, 0, stream>>>(z3, zu3, zn3, 3136, 3136, 128);
  k_zunit<<<1, 256, 0, stream>>>(z4, zu4, zn4, 1152, 1152, 128);
  k_zunit<<<2, 256, 0, stream>>>(zf1, zuf1, znf1, 352, 352, 512);
  k_zunit<<<4, 256, 0, stream>>>(zf2, zuf2, znf2, 512, 512, 1000);

  struct Layer {
    const float* in; int nchw; int B, H, W, Cin, KH, pad, K, Kp, N;
    float scale; const _Float16* zu; const float* zn; const float* bias;
    int* tab; float* pool_out;
  };
  Layer L[4] = {
    { x,   1, 16, 128, 128,   3, 3, 1,   27,   32,  32, sc1, zu1, zn1, b1, t1, x1p },
    { x1p, 0, 16,  64,  64,  32, 5, 2,  800,  800,  64, sc2, zu2, zn2, b2, t2, x2p },
    { x2p, 0, 16,  32,  32,  64, 7, 3, 3136, 3136, 128, sc3, zu3, zn3, b3, t3, x3p },
    { x3p, 0, 16,  16,  16, 128, 3, 1, 1152, 1152, 128, sc4, zu4, zn4, b4, t4, x4p },
  };
  for (int l = 0; l < 4; ++l) {
    const Layer& q = L[l];
    const int M  = q.B * q.H * q.W;
    const int Hp = q.H + 2 * q.pad, Wp = q.W + 2 * q.pad;
    const int nPad = q.B * Hp * Wp * q.Cin;
    k_mktab<<<(q.Kp + 255) / 256, 256, 0, stream>>>(q.tab, q.K, q.Kp, q.KH, Wp, q.Cin);
    k_fill<<<(nPad + 255) / 256, 256, 0, stream>>>(v, nPad);
    k_logmap<<<(M + 255) / 256, 256, 0, stream>>>(q.in, v, M, q.Cin, q.H, q.W,
                                                  q.pad, q.nchw);
    k_rowstats<<<(M + 255) / 256, 256, 0, stream>>>(v, rowscale, cx2, q.B, q.H, q.W,
                                                    q.Cin, q.KH, q.pad, q.scale);
    k_conv_gemm<<<dim3(M / 128, q.N / 32), 256, 0, stream>>>(
        v, rowscale, q.zu, q.tab, P, q.H, q.W, Hp, Wp, q.Cin, q.Kp, q.N);
    k_epilogue<<<M / 8, 256, 0, stream>>>(P, cx2, q.zn, q.bias, M, q.N);
    k_maxpool<<<(q.B * (q.H / 2) * (q.W / 2)) / 8, 256, 0, stream>>>(
        P, q.pool_out, q.B, q.H, q.W, q.N);
  }

  k_flat<<<(16 * 352 + 255) / 256, 256, 0, stream>>>(x1p, x2p, x3p, x4p, flat);
  k_fc<<<16, 256, 0, stream>>>(flat, zuf1, znf1, bf1, hbuf, 352, 512, 1);
  k_fc<<<16, 256, 0, stream>>>(hbuf, zuf2, znf2, bf2, (float*)d_out, 512, 1000, 0);
}
